// TransformerEncoderBlock_45896020525382
// MI455X (gfx1250) — matrix operand order repaired and checked by reading, not compiled, not run on hardware
//
#include <hip/hip_runtime.h>
#include <hip/hip_bf16.h>
#include <math.h>

typedef __attribute__((ext_vector_type(16))) __bf16 bf16x16;
typedef __attribute__((ext_vector_type(8)))  __bf16 bf16x8;
typedef __attribute__((ext_vector_type(4)))  __bf16 bf16x4;
typedef __attribute__((ext_vector_type(8)))  float  f32x8;

#define E_DIM 768
#define F_DIM 3072
#define SEQ   1024
#define NBATCH 8
#define ROWS  (NBATCH * SEQ)   // 8192
#define HEADS 8
#define HD    96

// ---------------------------------------------------------------------------
// WMMA helpers (gfx1250, wave32, 16x16x32 bf16 -> f32)
// ---------------------------------------------------------------------------
__device__ __forceinline__ f32x8 wmma_bf16(bf16x16 a, bf16x16 b, f32x8 c) {
  return __builtin_amdgcn_wmma_f32_16x16x32_bf16(
      /*neg_a=*/false, a, /*neg_b=*/false, b,
      /*c_mod=*/(short)0, c, /*reuse_a=*/false, /*reuse_b=*/false);
}

// A-matrix fragment, 16(M) x 32(K) bf16, from a row-major (m,k) tile.
// ISA: lanes 0-15 (M=lane) hold K=0..7 then 16..23; lanes 16-31 hold
// K=8..15 then 24..31.
__device__ __forceinline__ bf16x16 frag_a(const __bf16* base, int stride, int lane) {
  const int r = lane & 15, half = lane >> 4;
  const __bf16* p = base + r * stride + half * 8;
  const bf16x8 lo = *(const bf16x8*)(p);
  const bf16x8 hi = *(const bf16x8*)(p + 16);
  bf16x16 f;
#pragma unroll
  for (int i = 0; i < 8; ++i) { f[i] = lo[i]; f[i + 8] = hi[i]; }
  return f;
}

// B-matrix fragment, 32(K) x 16(N) bf16, from an N-major (n,k) tile.
// ISA: lanes 0-15 (N=lane) hold K=0..7 then 16..23; lanes 16-31 hold
// K=8..15 then 24..31.
__device__ __forceinline__ bf16x16 frag_b(const __bf16* base, int stride, int lane) {
  const int r = lane & 15, half = lane >> 4;
  const __bf16* p = base + r * stride + half * 8;
  const bf16x8 lo = *(const bf16x8*)(p);
  const bf16x8 hi = *(const bf16x8*)(p + 16);
  bf16x16 f;
#pragma unroll
  for (int i = 0; i < 8; ++i) { f[i] = lo[i]; f[i + 8] = hi[i]; }
  return f;
}

// ---------------------------------------------------------------------------
// Tiled transpose + f32->bf16 convert: in[K][N] f32  ->  out[N][K] bf16.
// 64x64 tiles through LDS so both the read and the write are coalesced.
// ---------------------------------------------------------------------------
__global__ __launch_bounds__(256) void cvtT_bf16_k(const float* __restrict__ in,
                                                   __bf16* __restrict__ out,
                                                   int K, int N) {
  __shared__ __bf16 t[64][68];   // pitch 68 elems = 136B (8B aligned rows)
  const int bk = blockIdx.y * 64, bn = blockIdx.x * 64;
  const int tid = threadIdx.x;
#pragma unroll
  for (int p = 0; p < 4; ++p) {
    int lin4 = p * 256 + tid;             // 0..1023 float4 slots
    int r = lin4 >> 4, c = (lin4 & 15) * 4;
    float4 v = *(const float4*)&in[(size_t)(bk + r) * N + bn + c];
    bf16x4 o;
    o[0] = (__bf16)v.x; o[1] = (__bf16)v.y; o[2] = (__bf16)v.z; o[3] = (__bf16)v.w;
    *(bf16x4*)&t[r][c] = o;
  }
  __syncthreads();
#pragma unroll
  for (int p = 0; p < 4; ++p) {
    int lin4 = p * 256 + tid;
    int r2 = lin4 >> 4, c2 = (lin4 & 15) * 4;
    bf16x4 o;
#pragma unroll
    for (int i = 0; i < 4; ++i) o[i] = t[c2 + i][r2];
    *(bf16x4*)&out[(size_t)(bn + r2) * K + bk + c2] = o;
  }
}

// ---------------------------------------------------------------------------
// LayerNorm (row = 768), f32 in -> bf16 out. One block per row, 256 threads.
// ---------------------------------------------------------------------------
__global__ __launch_bounds__(256) void ln_bf16_k(const float* __restrict__ x,
                                                 const float* __restrict__ g,
                                                 const float* __restrict__ beta,
                                                 __bf16* __restrict__ out) {
  __shared__ float red[16];
  const int row = blockIdx.x;
  const int t = threadIdx.x;
  const float* xr = x + (size_t)row * E_DIM;
  float a0 = xr[t], a1 = xr[t + 256], a2 = xr[t + 512];
  float s = a0 + a1 + a2;
  float sq = a0 * a0 + a1 * a1 + a2 * a2;
#pragma unroll
  for (int m = 1; m < 32; m <<= 1) { s += __shfl_xor(s, m); sq += __shfl_xor(sq, m); }
  const int wave = t >> 5, lane = t & 31;
  if (lane == 0) { red[wave] = s; red[8 + wave] = sq; }
  __syncthreads();
  float ts = 0.f, tq = 0.f;
#pragma unroll
  for (int i = 0; i < 8; ++i) { ts += red[i]; tq += red[8 + i]; }
  const float mean = ts * (1.0f / E_DIM);
  const float var  = tq * (1.0f / E_DIM) - mean * mean;
  const float rstd = rsqrtf(var + 1e-5f);
  __bf16* orow = out + (size_t)row * E_DIM;
  orow[t]       = (__bf16)((a0 - mean) * rstd * g[t]       + beta[t]);
  orow[t + 256] = (__bf16)((a1 - mean) * rstd * g[t + 256] + beta[t + 256]);
  orow[t + 512] = (__bf16)((a2 - mean) * rstd * g[t + 512] + beta[t + 512]);
}

// ---------------------------------------------------------------------------
// Tiled WMMA GEMM: C[M,N] = A[M,K](bf16,row-major) * WT[N,K](bf16,N-major).
// Block tile 128x256, K-step 32, 256 threads = 8 waves (2Mx4N grid), each
// wave owns a 64x64 patch (4x4 wmma tiles -> 16 WMMAs per 8 fragment loads
// per K-step). Next K-tile is prefetched into registers while the current
// LDS tile is consumed. Epilogues:
//   0: +bias            -> bf16            (Q/K projections)
//   1: +bias +residual  -> f32             (Wo projection + residual)
//   2: +bias, GELU      -> bf16            (FFN up-proj)
//   3: +bias            -> f32             (FFN down-proj, final output)
//   4: +bias            -> bf16 transposed per batch: vt[b][col][seq]
// ---------------------------------------------------------------------------
#define LDP 40   // LDS row pitch (elems): 80B, keeps 16B alignment for frags

template <int EPI>
__global__ __launch_bounds__(256) void gemm_bf16_k(
    const __bf16* __restrict__ A, const __bf16* __restrict__ WT,
    const float* __restrict__ bias, const float* __restrict__ resid,
    float* __restrict__ outF, __bf16* __restrict__ outH,
    int M, int N, int K) {
  __shared__ __bf16 As[128][LDP];   // row-major (m, k)
  __shared__ __bf16 Bs[256][LDP];   // N-major   (n, k)

  const int bm = blockIdx.y * 128, bn = blockIdx.x * 256;
  const int tid = threadIdx.x, lane = tid & 31, wave = tid >> 5;
  const int wm = (wave >> 2) * 64;   // 0 or 64
  const int wn = (wave & 3) * 64;    // 0,64,128,192

  // Staging coordinates: 16B per thread per pass; A = 2 passes, B = 4 passes.
  const int sr = tid >> 2;            // 0..63
  const int sc = (tid & 3) * 8;       // 0,8,16,24

  f32x8 acc[4][4] = {};
  bf16x8 pa[2], pb[4];

  // Prefetch first K-tile.
#pragma unroll
  for (int p = 0; p < 2; ++p)
    pa[p] = *(const bf16x8*)&A[(size_t)(bm + p * 64 + sr) * K + sc];
#pragma unroll
  for (int p = 0; p < 4; ++p)
    pb[p] = *(const bf16x8*)&WT[(size_t)(bn + p * 64 + sr) * K + sc];

  for (int k0 = 0; k0 < K; k0 += 32) {
    __syncthreads();
#pragma unroll
    for (int p = 0; p < 2; ++p) *(bf16x8*)&As[p * 64 + sr][sc] = pa[p];
#pragma unroll
    for (int p = 0; p < 4; ++p) *(bf16x8*)&Bs[p * 64 + sr][sc] = pb[p];
    __syncthreads();

    if (k0 + 32 < K) {
      const int kn = k0 + 32;
#pragma unroll
      for (int p = 0; p < 2; ++p)
        pa[p] = *(const bf16x8*)&A[(size_t)(bm + p * 64 + sr) * K + kn + sc];
#pragma unroll
      for (int p = 0; p < 4; ++p)
        pb[p] = *(const bf16x8*)&WT[(size_t)(bn + p * 64 + sr) * K + kn + sc];
    }

    bf16x16 af[4], bfr[4];
#pragma unroll
    for (int mt = 0; mt < 4; ++mt) af[mt] = frag_a(&As[wm + mt * 16][0], LDP, lane);
#pragma unroll
    for (int nt = 0; nt < 4; ++nt) bfr[nt] = frag_b(&Bs[wn + nt * 16][0], LDP, lane);
#pragma unroll
    for (int mt = 0; mt < 4; ++mt)
#pragma unroll
      for (int nt = 0; nt < 4; ++nt)
        acc[mt][nt] = wmma_bf16(af[mt], bfr[nt], acc[mt][nt]);
  }

  // Epilogue. C layout: VGPR v holds (row = v + 8*(lane>>4), col = lane&15).
  const int half = lane >> 4, r16 = lane & 15;
#pragma unroll
  for (int mt = 0; mt < 4; ++mt)
#pragma unroll
    for (int nt = 0; nt < 4; ++nt)
#pragma unroll
      for (int v = 0; v < 8; ++v) {
        const int row = bm + wm + mt * 16 + v + 8 * half;
        const int col = bn + wn + nt * 16 + r16;
        float val = acc[mt][nt][v] + bias[col];
        const size_t idx = (size_t)row * N + col;
        if (EPI == 0) {
          outH[idx] = (__bf16)val;
        } else if (EPI == 1) {
          outF[idx] = val + resid[idx];
        } else if (EPI == 2) {
          val = 0.5f * val * (1.0f + erff(val * 0.70710678118654752f));
          outH[idx] = (__bf16)val;
        } else if (EPI == 3) {
          outF[idx] = val;
        } else {  // EPI == 4: V stored transposed per batch: vt[b][col][seq]
          const int bb = row >> 10, seq = row & (SEQ - 1);
          outH[((size_t)bb * E_DIM + col) * SEQ + seq] = (__bf16)val;
        }
      }
}

// ---------------------------------------------------------------------------
// Flash-style attention: energy = q k^T (no pre-scale), softmax, *HD, @v.
// Grid: (B*H, SEQ/128). 256 threads = 8 fully independent waves, 16 query
// rows each. K (token-major) and Vt (dim-major) are already in B-fragment
// layout, so fragments are read straight from global (192MB L2-resident);
// no cross-wave barriers. Only per-wave LDS: P reshape buffer.
// ---------------------------------------------------------------------------
__global__ __launch_bounds__(256) void attn_k(const __bf16* __restrict__ q,
                                              const __bf16* __restrict__ k,
                                              const __bf16* __restrict__ vt,
                                              __bf16* __restrict__ o) {
  __shared__ __bf16 Pb[8][16][64];     // per-wave probs (q-row, key)

  const int bh = blockIdx.x;           // 0..63
  const int b  = bh >> 3, h = bh & 7;
  const int qblk = blockIdx.y * 128;
  const int tid = threadIdx.x, lane = tid & 31, wave = tid >> 5;
  const int half = lane >> 4, r16 = lane & 15;

  const size_t base  = ((size_t)b * SEQ) * E_DIM + h * HD;          // q,k,o
  const size_t baseV = ((size_t)b * E_DIM + h * HD) * SEQ;          // vt

  // Q fragments for this wave's 16 rows, kept in registers all kernel.
  bf16x16 qf[3];
  {
    const __bf16* qp = q + base + (size_t)(qblk + wave * 16) * E_DIM;
#pragma unroll
    for (int ks = 0; ks < 3; ++ks) qf[ks] = frag_a(qp + ks * 32, E_DIM, lane);
  }

  f32x8 Oa[6] = {};
  float mrow[8], lrow[8];
#pragma unroll
  for (int i = 0; i < 8; ++i) { mrow[i] = -1e30f; lrow[i] = 0.f; }

  for (int kc = 0; kc < SEQ; kc += 64) {
    // S = Q @ K^T: 16 queries x 64 keys; B-frags straight from global k.
    f32x8 S[4] = {};
#pragma unroll
    for (int nt = 0; nt < 4; ++nt)
#pragma unroll
      for (int ks = 0; ks < 3; ++ks)
        S[nt] = wmma_bf16(
            qf[ks],
            frag_b(k + base + (size_t)(kc + nt * 16) * E_DIM + ks * 32, E_DIM, lane),
            S[nt]);

    // Online softmax (per row = vv + 8*half; stats redundant across 16 lanes).
#pragma unroll
    for (int vv = 0; vv < 8; ++vv) {
      float mx = fmaxf(fmaxf(S[0][vv], S[1][vv]), fmaxf(S[2][vv], S[3][vv]));
      mx = fmaxf(mx, __shfl_xor(mx, 1));
      mx = fmaxf(mx, __shfl_xor(mx, 2));
      mx = fmaxf(mx, __shfl_xor(mx, 4));
      mx = fmaxf(mx, __shfl_xor(mx, 8));
      const float mn = fmaxf(mrow[vv], mx);
      const float corr = __expf(mrow[vv] - mn);
      mrow[vv] = mn;
      float rs = 0.f;
#pragma unroll
      for (int nt = 0; nt < 4; ++nt) {
        const float pv = __expf(S[nt][vv] - mn);
        rs += pv;
        Pb[wave][vv + 8 * half][nt * 16 + r16] = (__bf16)pv;
      }
      rs += __shfl_xor(rs, 1);
      rs += __shfl_xor(rs, 2);
      rs += __shfl_xor(rs, 4);
      rs += __shfl_xor(rs, 8);
      lrow[vv] = lrow[vv] * corr + rs;
#pragma unroll
      for (int t = 0; t < 6; ++t) Oa[t][vv] *= corr;
    }
    // Wave-local LDS visibility before re-reading Pb as A-fragments.
    asm volatile("s_wait_dscnt 0x0" ::: "memory");

    // O += P @ V: B-frags straight from global vt (dim-major).
#pragma unroll
    for (int t = 0; t < 6; ++t)
#pragma unroll
      for (int ks = 0; ks < 2; ++ks)
        Oa[t] = wmma_bf16(
            frag_a(&Pb[wave][0][ks * 32], 64, lane),
            frag_b(vt + baseV + (size_t)(t * 16) * SEQ + kc + ks * 32, SEQ, lane),
            Oa[t]);
  }

  // Final: out = (softmax * HD) @ v  ->  scale by HD / l.
#pragma unroll
  for (int vv = 0; vv < 8; ++vv) {
    const float sc = (float)HD / lrow[vv];
    const int row = qblk + wave * 16 + vv + 8 * half;
#pragma unroll
    for (int t = 0; t < 6; ++t)
      o[base + (size_t)row * E_DIM + t * 16 + r16] = (__bf16)(Oa[t][vv] * sc);
  }
}

// ---------------------------------------------------------------------------
// Host-side orchestration
// ---------------------------------------------------------------------------
extern "C" void kernel_launch(void* const* d_in, const int* in_sizes, int n_in,
                              void* d_out, int out_size, void* d_ws, size_t ws_size,
                              hipStream_t stream) {
  (void)in_sizes; (void)n_in; (void)out_size; (void)ws_size;
  const float* x    = (const float*)d_in[0];
  const float* g1   = (const float*)d_in[1];
  const float* be1  = (const float*)d_in[2];
  const float* Wq   = (const float*)d_in[3];
  const float* bq   = (const float*)d_in[4];
  const float* Wk   = (const float*)d_in[5];
  const float* bk   = (const float*)d_in[6];
  const float* Wv   = (const float*)d_in[7];
  const float* bv   = (const float*)d_in[8];
  const float* Wo   = (const float*)d_in[9];
  const float* bo   = (const float*)d_in[10];
  const float* g2   = (const float*)d_in[11];
  const float* be2  = (const float*)d_in[12];
  const float* W1   = (const float*)d_in[13];
  const float* b1   = (const float*)d_in[14];
  const float* W2   = (const float*)d_in[15];
  const float* b2   = (const float*)d_in[16];

  char* ws = (char*)d_ws;
  size_t off = 0;
  auto alloc = [&](size_t bytes) -> void* {
    void* p = ws + off;
    off += (bytes + 255) & ~(size_t)255;
    return p;
  };

  __bf16* hb   = (__bf16*)alloc((size_t)ROWS * E_DIM * 2);  // LN1 out
  __bf16* qb   = (__bf16*)alloc((size_t)ROWS * E_DIM * 2);
  __bf16* kb   = (__bf16*)alloc((size_t)ROWS * E_DIM * 2);
  __bf16* vtb  = (__bf16*)alloc((size_t)ROWS * E_DIM * 2);  // V, [b][dim][seq]
  __bf16* ab   = (__bf16*)alloc((size_t)ROWS * E_DIM * 2);  // attention out
  float*  resf = (float*) alloc((size_t)ROWS * E_DIM * 4);  // x + attn proj
  __bf16* h2b  = (__bf16*)alloc((size_t)ROWS * E_DIM * 2);  // LN2 out
  __bf16* fb   = (__bf16*)alloc((size_t)ROWS * F_DIM * 2);  // FFN1 out
  __bf16* WqT  = (__bf16*)alloc((size_t)E_DIM * E_DIM * 2); // all weights
  __bf16* WkT  = (__bf16*)alloc((size_t)E_DIM * E_DIM * 2); //   pre-transposed
  __bf16* WvT  = (__bf16*)alloc((size_t)E_DIM * E_DIM * 2); //   to [N][K] bf16
  __bf16* WoT  = (__bf16*)alloc((size_t)E_DIM * E_DIM * 2);
  __bf16* W1T  = (__bf16*)alloc((size_t)E_DIM * F_DIM * 2);
  __bf16* W2T  = (__bf16*)alloc((size_t)F_DIM * E_DIM * 2);

  const dim3 blk(256);

  // Weight transpose + bf16 conversion.
  const dim3 gEE(E_DIM / 64, E_DIM / 64);
  cvtT_bf16_k<<<gEE, blk, 0, stream>>>(Wq, WqT, E_DIM, E_DIM);
  cvtT_bf16_k<<<gEE, blk, 0, stream>>>(Wk, WkT, E_DIM, E_DIM);
  cvtT_bf16_k<<<gEE, blk, 0, stream>>>(Wv, WvT, E_DIM, E_DIM);
  cvtT_bf16_k<<<gEE, blk, 0, stream>>>(Wo, WoT, E_DIM, E_DIM);
  cvtT_bf16_k<<<dim3(F_DIM / 64, E_DIM / 64), blk, 0, stream>>>(W1, W1T, E_DIM, F_DIM);
  cvtT_bf16_k<<<dim3(E_DIM / 64, F_DIM / 64), blk, 0, stream>>>(W2, W2T, F_DIM, E_DIM);

  // LN1.
  ln_bf16_k<<<ROWS, blk, 0, stream>>>(x, g1, be1, hb);

  // Q,K projections (token-major bf16); V projection stored dim-major.
  const dim3 gE(E_DIM / 256, ROWS / 128);
  gemm_bf16_k<0><<<gE, blk, 0, stream>>>(hb, WqT, bq, nullptr, nullptr, qb,
                                         ROWS, E_DIM, E_DIM);
  gemm_bf16_k<0><<<gE, blk, 0, stream>>>(hb, WkT, bk, nullptr, nullptr, kb,
                                         ROWS, E_DIM, E_DIM);
  gemm_bf16_k<4><<<gE, blk, 0, stream>>>(hb, WvT, bv, nullptr, nullptr, vtb,
                                         ROWS, E_DIM, E_DIM);

  // Attention (flash-style, post-softmax *HD folded into normalization).
  attn_k<<<dim3(NBATCH * HEADS, SEQ / 128), blk, 0, stream>>>(qb, kb, vtb, ab);

  // Output projection + residual.
  gemm_bf16_k<1><<<gE, blk, 0, stream>>>(ab, WoT, bo, x, resf, nullptr,
                                         ROWS, E_DIM, E_DIM);

  // LN2.
  ln_bf16_k<<<ROWS, blk, 0, stream>>>(resf, g2, be2, h2b);

  // FFN: up-proj + exact GELU, then down-proj to f32 output.
  const dim3 gF(F_DIM / 256, ROWS / 128);
  gemm_bf16_k<2><<<gF, blk, 0, stream>>>(h2b, W1T, b1, nullptr, nullptr, fb,
                                         ROWS, F_DIM, E_DIM);
  gemm_bf16_k<3><<<gE, blk, 0, stream>>>(fb, W2T, b2, nullptr, (float*)d_out,
                                         nullptr, ROWS, E_DIM, F_DIM);
}
